// SAINTReconstructionHead_64055142253058
// MI455X (gfx1250) — matrix operand order repaired and checked by reading, not compiled, not run on hardware
//
#include <hip/hip_runtime.h>

// ---------------------------------------------------------------------------
// SAINT reconstruction head, fused per-feature 2-layer MLP on CDNA5 (gfx1250).
//
// Compute-bound (~590 FLOP/byte) => bf16 WMMA 16x16x32 (f32 accum).
// v2: pre-convert W1/W2 to bf16 in workspace, then stream weight panels into
// LDS with double-buffered GLOBAL_LOAD_ASYNC_TO_LDS_B128 (ASYNCcnt), so the
// DMA overlaps the WMMA stream instead of sitting between barriers.
// ---------------------------------------------------------------------------

typedef __attribute__((ext_vector_type(16))) __bf16 v16bf;
typedef __attribute__((ext_vector_type(8)))  float  v8f;

#define B_   2048
#define F_   32
#define D_   512
#define H_   2560
#define OMAX 40
#define OPAD 64
#define TOTAL_OUT 255

#define BM 128   // batch rows per workgroup
#define HN 64    // H chunk width (2560/64 = 40 chunks)
#define NCHUNK (H_ / HN)
#define XS 520   // xA  LDS row stride in halfs (512 + 8 pad)
#define KS 520   // w1t LDS row stride in halfs (K-major, 512 + 8 pad)
#define HS 72    // h_s LDS row stride in halfs (64 + 8 pad)
#define WS 72    // w2t LDS row stride in halfs (64 + 8 pad)

// LDS byte-offset map (dynamic LDS base == 0; no static __shared__ in kernel)
#define XA_OFF   0u
#define XA_SZ    (BM * XS * 2u)                 // 133120
#define W1T_SZ   (HN * KS * 2u)                 // 66560
#define W1T_OFF(p) (XA_SZ + (p) * W1T_SZ)
#define HS_OFF   (XA_SZ + 2u * W1T_SZ)          // 266240
#define HS_SZ    (BM * HS * 2u)                 // 18432
#define W2T_SZ   (OPAD * WS * 2u)               // 9216
#define W2T_OFF(p) (HS_OFF + HS_SZ + (p) * W2T_SZ)
#define LDS_TOTAL (HS_OFF + HS_SZ + 2u * W2T_SZ)   // 303104 B <= 320KB

__constant__ int c_odim[F_] = {12,1,7,1,25,3,1,9,1,15,2,1,40,1,5,8,1,11,1,6,
                               30,1,4,1,18,1,10,2,1,22,1,13};
__constant__ int c_coff[F_] = {0,12,13,20,21,46,49,50,59,60,75,77,78,118,119,
                               124,132,133,144,145,151,181,182,186,187,205,206,
                               216,218,219,241,242};

__device__ __forceinline__ unsigned short f2bf(float f) {
  __bf16 b = (__bf16)f;                      // hardware cvt, RNE
  return __builtin_bit_cast(unsigned short, b);
}

__device__ __forceinline__ void wait_async0() {
#if __has_builtin(__builtin_amdgcn_s_wait_asynccnt)
  __builtin_amdgcn_s_wait_asynccnt(0);
#else
  asm volatile("s_wait_asynccnt 0x0" ::: "memory");
#endif
}

__device__ __forceinline__ void async_b128(unsigned lds_byte_off, const void* gaddr) {
  // per-lane 16B copy, tracked by ASYNCcnt
  asm volatile("global_load_async_to_lds_b128 %0, %1, off"
               :: "v"(lds_byte_off), "v"(gaddr) : "memory");
}

union FragU { uint4 q[2]; v16bf v; };

// ---------------------------------------------------------------------------
// Pre-pass 1: W1 [f][d][h] f32  ->  ws [f][h][d] bf16  (LDS tile transpose)
// ---------------------------------------------------------------------------
__global__ __launch_bounds__(256)
void conv_w1_kernel(const float* __restrict__ W1, unsigned short* __restrict__ w1bf) {
  __shared__ float tile[64][65];
  const int tid = threadIdx.x;
  const int hblk = blockIdx.x * 64;   // H/64 = 40
  const int dblk = blockIdx.y * 64;   // D/64 = 8
  const int f    = blockIdx.z;        // F  = 32
  const float* src = W1 + ((size_t)f * D_ + dblk) * H_ + hblk;
  #pragma unroll
  for (int p = 0; p < 4; ++p) {
    int d  = p * 16 + (tid >> 4);
    int c4 = (tid & 15) * 4;
    float4 v = *(const float4*)(src + (size_t)d * H_ + c4);
    tile[d][c4 + 0] = v.x; tile[d][c4 + 1] = v.y;
    tile[d][c4 + 2] = v.z; tile[d][c4 + 3] = v.w;
  }
  __syncthreads();
  unsigned short* dst = w1bf + ((size_t)f * H_ + hblk) * D_ + dblk;
  #pragma unroll
  for (int p = 0; p < 4; ++p) {
    int h  = p * 16 + (tid >> 4);
    int c4 = (tid & 15) * 4;
    ushort4 o;
    o.x = f2bf(tile[c4 + 0][h]); o.y = f2bf(tile[c4 + 1][h]);
    o.z = f2bf(tile[c4 + 2][h]); o.w = f2bf(tile[c4 + 3][h]);
    *(ushort4*)(dst + (size_t)h * D_ + c4) = o;
  }
}

// ---------------------------------------------------------------------------
// Pre-pass 2: W2 [f][h][40] f32 -> ws [f][64][h] bf16 (transpose + zero pad)
// ---------------------------------------------------------------------------
__global__ __launch_bounds__(256)
void conv_w2_kernel(const float* __restrict__ W2, unsigned short* __restrict__ w2bf) {
  size_t idx = (size_t)blockIdx.x * 256 + threadIdx.x;   // F*OPAD*H elements
  if (idx >= (size_t)F_ * OPAD * H_) return;
  int h = (int)(idx % H_);
  int r = (int)(idx / H_);
  int o = r % OPAD;
  int f = r / OPAD;
  float v = (o < OMAX) ? W2[((size_t)f * H_ + h) * OMAX + o] : 0.0f;
  w2bf[idx] = f2bf(v);
}

// ---------------------------------------------------------------------------
// Main fused kernel, async double-buffered weight panels
// ---------------------------------------------------------------------------
__global__ __launch_bounds__(256)
void saint_head_async(const float* __restrict__ x,
                      const unsigned short* __restrict__ w1bf,
                      const float* __restrict__ b1,
                      const unsigned short* __restrict__ w2bf,
                      const float* __restrict__ b2,
                      float* __restrict__ out)
{
  extern __shared__ __align__(16) unsigned short smem[];
  char* sbase = (char*)smem;

  const int tid  = threadIdx.x;
  const int lane = tid & 31;
  const int wave = tid >> 5;        // 8 waves
  const int l16  = lane & 15;
  const int hi   = lane >> 4;
  const int b0   = blockIdx.x * BM;
  const int f    = blockIdx.y;

  // issue one chunk's weight panels into LDS buffer p (ASYNCcnt-tracked DMA)
  auto issue_panel = [&](int hc, int p) {
    const unsigned short* g1 = w1bf + ((size_t)f * H_ + hc) * D_;  // 64KB contiguous
    #pragma unroll
    for (int i = 0; i < 16; ++i) {
      int lin = tid + 256 * i;               // 0..4095 16B-chunks
      int h  = lin >> 6;                     // 64 chunks per 512-half row
      int dc = lin & 63;
      async_b128(W1T_OFF(p) + (unsigned)h * (KS * 2) + (unsigned)dc * 16,
                 g1 + (size_t)h * D_ + dc * 8);
    }
    const unsigned short* g2 = w2bf + (size_t)f * OPAD * H_ + hc;
    #pragma unroll
    for (int i = 0; i < 2; ++i) {
      int lin = tid + 256 * i;               // 0..511 16B-chunks
      int o = lin >> 3;                      // 8 chunks per 64-half row
      int c = lin & 7;
      async_b128(W2T_OFF(p) + (unsigned)o * (WS * 2) + (unsigned)c * 16,
                 g2 + (size_t)o * H_ + c * 8);
    }
  };

  // ---- prologue: start chunk-0 DMA, stage x tile (f32->bf16) meanwhile
  issue_panel(0, 0);
  {
    unsigned short* xA = (unsigned short*)(sbase + XA_OFF);
    const float* xbase = x + (size_t)b0 * (F_ * D_) + (size_t)f * D_;
    for (int idx = tid; idx < BM * (D_ / 4); idx += 256) {
      int row = idx >> 7;
      int c4  = idx & 127;
      float4 v = ((const float4*)(xbase + (size_t)row * (F_ * D_)))[c4];
      ushort4 pk;
      pk.x = f2bf(v.x); pk.y = f2bf(v.y); pk.z = f2bf(v.z); pk.w = f2bf(v.w);
      *(ushort4*)&xA[row * XS + c4 * 4] = pk;
    }
  }
  wait_async0();
  __syncthreads();

  v8f acc2[4] = {};
  const unsigned xa_row = XA_OFF + (unsigned)(wave * 16 + l16) * (XS * 2);

  for (int ci = 0; ci < NCHUNK; ++ci) {
    const int hc = ci * HN;
    const int p  = ci & 1;
    if (ci + 1 < NCHUNK) issue_panel(hc + HN, p ^ 1);   // overlap DMA with WMMA

    // ---- GEMM1: rows [16w,16w+16) x cols [0,64), K = 512
    v8f acc1[4] = {};
    for (int kk = 0; kk < D_ / 32; ++kk) {
      FragU A;  // lane = row M; K = 32kk+16hi+0..15 contiguous
      const char* pa = sbase + xa_row + kk * 64 + hi * 32;
      A.q[0] = *(const uint4*)pa;
      A.q[1] = *(const uint4*)(pa + 16);
      #pragma unroll
      for (int n = 0; n < 4; ++n) {
        FragU Bf;  // lane = col N; K = 32kk+16hi+0..15 contiguous
        const char* pb = sbase + W1T_OFF(p)
                       + (unsigned)(n * 16 + l16) * (KS * 2) + kk * 64 + hi * 32;
        Bf.q[0] = *(const uint4*)pb;
        Bf.q[1] = *(const uint4*)(pb + 16);
        acc1[n] = __builtin_amdgcn_wmma_f32_16x16x32_bf16(
            false, A.v, false, Bf.v, (short)0, acc1[n], false, false);
      }
    }

    // ---- bias + ReLU -> bf16 -> per-wave-private rows of hS
    unsigned short* hS = (unsigned short*)(sbase + HS_OFF);
    #pragma unroll
    for (int n = 0; n < 4; ++n) {
      int col = n * 16 + l16;
      float bv = b1[(size_t)f * H_ + hc + col];
      #pragma unroll
      for (int r = 0; r < 8; ++r) {
        float v = acc1[n][r] + bv;          // C layout: vgpr r -> row r+8*hi
        v = v > 0.0f ? v : 0.0f;
        hS[(wave * 16 + r + hi * 8) * HS + col] = f2bf(v);
      }
    }
    // same wave produces & consumes its hS rows: DScnt waits suffice, no barrier

    // ---- GEMM2: acc2 += h_chunk @ W2_chunk (K = 64)
    #pragma unroll
    for (int kk = 0; kk < HN / 32; ++kk) {
      FragU A;
      const char* pa = sbase + HS_OFF
                     + (unsigned)(wave * 16 + l16) * (HS * 2) + kk * 64 + hi * 32;
      A.q[0] = *(const uint4*)pa;
      A.q[1] = *(const uint4*)(pa + 16);
      #pragma unroll
      for (int n = 0; n < 4; ++n) {
        FragU Bf;
        const char* pb = sbase + W2T_OFF(p)
                       + (unsigned)(n * 16 + l16) * (WS * 2) + kk * 64 + hi * 32;
        Bf.q[0] = *(const uint4*)pb;
        Bf.q[1] = *(const uint4*)(pb + 16);
        acc2[n] = __builtin_amdgcn_wmma_f32_16x16x32_bf16(
            false, A.v, false, Bf.v, (short)0, acc2[n], false, false);
      }
    }

    wait_async0();      // next panel fully in LDS (own-wave ASYNCcnt)
    __syncthreads();    // all waves done reading buffer p before it is reused
  }

  // ---- epilogue: + b2, scatter this feature's valid columns
  const int odim = c_odim[f];
  const int coff = c_coff[f];
  #pragma unroll
  for (int n = 0; n < 4; ++n) {
    int o = n * 16 + l16;
    if (o < odim) {
      float b2v = b2[f * OMAX + o];
      #pragma unroll
      for (int r = 0; r < 8; ++r) {
        int row = b0 + wave * 16 + r + hi * 8;
        out[(size_t)row * TOTAL_OUT + coff + o] = acc2[n][r] + b2v;
      }
    }
  }
}

// ---------------------------------------------------------------------------
// Fallback (ws too small): round-1 fused kernel with inline conversion
// ---------------------------------------------------------------------------
__global__ __launch_bounds__(256)
void saint_head_inline(const float* __restrict__ x,  const float* __restrict__ W1,
                       const float* __restrict__ b1, const float* __restrict__ W2,
                       const float* __restrict__ b2, float* __restrict__ out)
{
  extern __shared__ __align__(16) unsigned short smem[];
  unsigned short* xA  = smem;
  unsigned short* w1t = xA  + BM * XS;
  unsigned short* hS  = w1t + HN * KS;
  unsigned short* w2t = hS  + BM * HS;

  const int tid  = threadIdx.x;
  const int lane = tid & 31;
  const int wave = tid >> 5;
  const int l16  = lane & 15;
  const int hi   = lane >> 4;
  const int b0   = blockIdx.x * BM;
  const int f    = blockIdx.y;

  {
    const float* xbase = x + (size_t)b0 * (F_ * D_) + (size_t)f * D_;
    for (int idx = tid; idx < BM * (D_ / 4); idx += 256) {
      int row = idx >> 7, c4 = idx & 127;
      float4 v = ((const float4*)(xbase + (size_t)row * (F_ * D_)))[c4];
      ushort4 pk;
      pk.x = f2bf(v.x); pk.y = f2bf(v.y); pk.z = f2bf(v.z); pk.w = f2bf(v.w);
      *(ushort4*)&xA[row * XS + c4 * 4] = pk;
    }
  }
  v8f acc2[4] = {};
  __syncthreads();

  for (int hc = 0; hc < H_; hc += HN) {
    const float* w1base = W1 + (size_t)f * D_ * H_ + hc;
    for (int it = 0; it < D_ / 16; ++it) {
      int d = it * 16 + (tid >> 4);
      int c = (tid & 15) * 4;
      float4 v = *(const float4*)(w1base + (size_t)d * H_ + c);
      w1t[(c + 0) * KS + d] = f2bf(v.x);
      w1t[(c + 1) * KS + d] = f2bf(v.y);
      w1t[(c + 2) * KS + d] = f2bf(v.z);
      w1t[(c + 3) * KS + d] = f2bf(v.w);
    }
    {
      int k = tid >> 2, q = tid & 3;
      const float* w2base = W2 + ((size_t)f * H_ + hc + k) * OMAX;
      #pragma unroll
      for (int j = 0; j < 16; ++j) {
        int o = q * 16 + j;
        w2t[o * WS + k] = f2bf((o < OMAX) ? w2base[o] : 0.0f);
      }
    }
    __syncthreads();

    v8f acc1[4] = {};
    const int mrow = wave * 16 + l16;
    for (int kk = 0; kk < D_ / 32; ++kk) {
      FragU A;
      const unsigned short* pa = &xA[mrow * XS + kk * 32 + hi * 16];
      A.q[0] = *(const uint4*)pa;
      A.q[1] = *(const uint4*)(pa + 8);
      #pragma unroll
      for (int n = 0; n < 4; ++n) {
        FragU Bf;
        const unsigned short* pb = &w1t[(n * 16 + l16) * KS + kk * 32 + hi * 16];
        Bf.q[0] = *(const uint4*)pb;
        Bf.q[1] = *(const uint4*)(pb + 8);
        acc1[n] = __builtin_amdgcn_wmma_f32_16x16x32_bf16(
            false, A.v, false, Bf.v, (short)0, acc1[n], false, false);
      }
    }
    #pragma unroll
    for (int n = 0; n < 4; ++n) {
      int col = n * 16 + l16;
      float bv = b1[(size_t)f * H_ + hc + col];
      #pragma unroll
      for (int r = 0; r < 8; ++r) {
        float v = acc1[n][r] + bv;
        v = v > 0.0f ? v : 0.0f;
        hS[(wave * 16 + r + hi * 8) * HS + col] = f2bf(v);
      }
    }
    #pragma unroll
    for (int kk = 0; kk < HN / 32; ++kk) {
      FragU A;
      const unsigned short* pa = &hS[(wave * 16 + l16) * HS + kk * 32 + hi * 16];
      A.q[0] = *(const uint4*)pa;
      A.q[1] = *(const uint4*)(pa + 8);
      #pragma unroll
      for (int n = 0; n < 4; ++n) {
        FragU Bf;
        const unsigned short* pb = &w2t[(n * 16 + l16) * WS + kk * 32 + hi * 16];
        Bf.q[0] = *(const uint4*)pb;
        Bf.q[1] = *(const uint4*)(pb + 8);
        acc2[n] = __builtin_amdgcn_wmma_f32_16x16x32_bf16(
            false, A.v, false, Bf.v, (short)0, acc2[n], false, false);
      }
    }
    __syncthreads();
  }

  const int odim = c_odim[f];
  const int coff = c_coff[f];
  #pragma unroll
  for (int n = 0; n < 4; ++n) {
    int o = n * 16 + l16;
    if (o < odim) {
      float b2v = b2[f * OMAX + o];
      #pragma unroll
      for (int r = 0; r < 8; ++r) {
        int row = b0 + wave * 16 + r + hi * 8;
        out[(size_t)row * TOTAL_OUT + coff + o] = acc2[n][r] + b2v;
      }
    }
  }
}

extern "C" void kernel_launch(void* const* d_in, const int* in_sizes, int n_in,
                              void* d_out, int out_size, void* d_ws, size_t ws_size,
                              hipStream_t stream) {
  (void)in_sizes; (void)n_in; (void)out_size;
  const float* x  = (const float*)d_in[0];
  const float* W1 = (const float*)d_in[1];
  const float* b1 = (const float*)d_in[2];
  const float* W2 = (const float*)d_in[3];
  const float* b2 = (const float*)d_in[4];
  float* out = (float*)d_out;

  const size_t w1_elems = (size_t)F_ * H_ * D_;      // 41.9M bf16
  const size_t w2_elems = (size_t)F_ * OPAD * H_;    // 5.24M bf16
  const size_t ws_need  = (w1_elems + w2_elems) * sizeof(unsigned short);  // ~94.4MB

  dim3 grid(B_ / BM, F_);   // 16 x 32 = 512 workgroups

  if (ws_size >= ws_need && d_ws != nullptr) {
    unsigned short* w1bf = (unsigned short*)d_ws;
    unsigned short* w2bf = w1bf + w1_elems;
    // pre-convert weights to bf16 (runs every call: deterministic)
    conv_w1_kernel<<<dim3(H_ / 64, D_ / 64, F_), 256, 0, stream>>>(W1, w1bf);
    conv_w2_kernel<<<(unsigned)((w2_elems + 255) / 256), 256, 0, stream>>>(W2, w2bf);
    saint_head_async<<<grid, 256, LDS_TOTAL, stream>>>(x, w1bf, b1, w2bf, b2, out);
  } else {
    size_t smem = (size_t)(BM * XS + HN * KS + BM * HS + OPAD * WS)
                * sizeof(unsigned short);
    saint_head_inline<<<grid, 256, smem, stream>>>(x, W1, b1, W2, b2, out);
  }
}
